// QuantizedLinear_67903432949777
// MI455X (gfx1250) — matrix operand order repaired and checked by reading, not compiled, not run on hardware
//
#include <hip/hip_runtime.h>

// ---------------------------------------------------------------------------
// int4 group-quantized GEMM for gfx1250 (MI455X):
//   out[M,N] = x[M,K] * dequant(Wpacked)[N,K]^T + bias
// M=8192, N=4096, K=4096, GROUP_SIZE=128.
// Compute-bound (~275 GFLOP vs ~302MB @ 23.3TB/s): dequantize to bf16 in LDS,
// drive v_wmma_f32_16x16x32_bf16 with f32 accumulation.
// ---------------------------------------------------------------------------

typedef __attribute__((ext_vector_type(16))) __bf16    v16bf;
typedef __attribute__((ext_vector_type(8)))  float     v8f;
typedef __attribute__((ext_vector_type(4)))  unsigned  uint4x;

#define GK        4096       // IN_FEATURES
#define GN        4096       // OUT_FEATURES
#define GM        8192       // BATCH*SEQ
#define NGROUPS   32         // GK / GROUP_SIZE(128)
#define BM        128
#define BN        128
#define BK        64
#define KTILES    (GK / BK)  // 64
#define PAIRS     (BK / 2)   // 32 bf16-pairs (uints) per row per tile

// round-to-nearest-even f32 -> bf16, packed pair (lo in [15:0], hi in [31:16])
static __device__ __forceinline__ unsigned f2bf2(float lo, float hi) {
    unsigned a = __builtin_bit_cast(unsigned, lo);
    unsigned b = __builtin_bit_cast(unsigned, hi);
    a = (a + 0x7fffu + ((a >> 16) & 1u)) >> 16;
    b = (b + 0x7fffu + ((b >> 16) & 1u)) >> 16;
    return (a & 0xffffu) | (b << 16);
}

union Frag {                 // 32 bytes = 16 bf16 = 8 VGPRs per lane
    uint4x q[2];
    v16bf  v;
};

__global__ __launch_bounds__(256)
void qlinear_wmma_kernel(const float* __restrict__ x,
                         const int*   __restrict__ wpacked,   // [GN, GK/2] one byte per int32
                         const float* __restrict__ scales,    // [GN, NGROUPS]
                         const float* __restrict__ bias,      // [GN]
                         float*       __restrict__ out)       // [GM, GN]
{
    extern __shared__ unsigned smem[];
    unsigned* xs = smem;                       // [2][BM][PAIRS] bf16 pairs of x
    unsigned* ws = smem + 2 * BM * PAIRS;      // [2][BN][PAIRS] bf16 pairs of W

    const int t    = threadIdx.x;
    const int m0   = blockIdx.y * BM;
    const int n0   = blockIdx.x * BN;
    const int tx   = t & 31;    // pair column within tile (k = 2*tx, 2*tx+1)
    const int ty   = t >> 5;    // 0..7 row phase
    const int lane = t & 31;
    const int wave = t >> 5;    // 8 waves
    const int wm   = wave >> 1; // 0..3 : 32-row strip
    const int wn   = wave & 1;  // 0..1 : 64-col strip
    const int khalf = lane >> 4;
    const int lrow  = lane & 15;

    v8f acc[2][4];
    {
        v8f zero = {};
        #pragma unroll
        for (int mi = 0; mi < 2; ++mi)
            #pragma unroll
            for (int ni = 0; ni < 4; ++ni)
                acc[mi][ni] = zero;
    }

    // ---- stage tile kt into LDS buffer `buf` ------------------------------
    auto stage = [&](int buf, int kt) {
        const int k0 = kt * BK;
        unsigned* xb = xs + buf * (BM * PAIRS);
        unsigned* wb = ws + buf * (BN * PAIRS);
        #pragma unroll
        for (int r = ty; r < BM; r += 8) {
            // activations: f32 -> bf16 pair
            const float2 v = *(const float2*)(x + (size_t)(m0 + r) * GK + k0 + 2 * tx);
            xb[r * PAIRS + tx] = f2bf2(v.x, v.y);
            // weights: one packed byte -> two int4 -> dequant -> bf16 pair
            const int   o  = n0 + r;
            const int   pb = wpacked[(size_t)o * (GK / 2) + (k0 >> 1) + tx];
            const float s  = scales[o * NGROUPS + (k0 >> 7)];
            const float lo = (float)((pb & 15) - 8) * s;
            const float hi = (float)(((pb >> 4) & 15) - 8) * s;
            wb[r * PAIRS + tx] = f2bf2(lo, hi);
        }
    };

    // ---- compute on LDS buffer `buf` --------------------------------------
    auto compute = [&](int buf) {
        unsigned* xb = xs + buf * (BM * PAIRS);
        unsigned* wb = ws + buf * (BN * PAIRS);
        #pragma unroll
        for (int s = 0; s < 2; ++s) {          // two K=32 steps per BK=64 tile
            Frag a[2], b[4];
            // A frag (16x32 bf16): lanes 0-15 -> K 0..7,16..23, lanes 16-31 -> K 8..15,24..31
            #pragma unroll
            for (int mi = 0; mi < 2; ++mi) {
                const int row = wm * 32 + mi * 16 + lrow;
                const unsigned* p = xb + row * PAIRS + 16 * s;
                a[mi].q[0] = *(const uint4x*)(p + 4 * khalf);
                a[mi].q[1] = *(const uint4x*)(p + 8 + 4 * khalf);
            }
            // B frag (32x16 bf16): column = W row; lanes 0-15 K 0..7,16..23; 16-31 K 8..15,24..31
            #pragma unroll
            for (int ni = 0; ni < 4; ++ni) {
                const int col = wn * 64 + ni * 16 + lrow;
                const unsigned* p = wb + col * PAIRS + 16 * s;
                b[ni].q[0] = *(const uint4x*)(p + 4 * khalf);
                b[ni].q[1] = *(const uint4x*)(p + 8 + 4 * khalf);
            }
            #pragma unroll
            for (int mi = 0; mi < 2; ++mi)
                #pragma unroll
                for (int ni = 0; ni < 4; ++ni)
                    acc[mi][ni] = __builtin_amdgcn_wmma_f32_16x16x32_bf16(
                        false, a[mi].v, false, b[ni].v,
                        (short)0, acc[mi][ni], false, false);
        }
    };

    // ---- main loop: double-buffered ---------------------------------------
    stage(0, 0);
    __syncthreads();
    for (int kt = 0; kt < KTILES; ++kt) {
        const int cur = kt & 1;
        if (kt + 1 < KTILES) stage(cur ^ 1, kt + 1);
        compute(cur);
        __syncthreads();
    }

    // ---- epilogue: C/D layout (VGPR r -> M = r + 8*khalf, N = lrow) + bias -
    #pragma unroll
    for (int mi = 0; mi < 2; ++mi) {
        #pragma unroll
        for (int ni = 0; ni < 4; ++ni) {
            const int   gcol = n0 + wn * 64 + ni * 16 + lrow;
            const float bv   = bias[gcol];
            #pragma unroll
            for (int r = 0; r < 8; ++r) {
                const int gm = m0 + wm * 32 + mi * 16 + khalf * 8 + r;
                out[(size_t)gm * GN + gcol] = acc[mi][ni][r] + bv;
            }
        }
    }
}

extern "C" void kernel_launch(void* const* d_in, const int* in_sizes, int n_in,
                              void* d_out, int out_size, void* d_ws, size_t ws_size,
                              hipStream_t stream) {
    const float* x       = (const float*)d_in[0];
    const int*   wpacked = (const int*)  d_in[1];
    const float* scales  = (const float*)d_in[2];
    // d_in[3] = zeros (unused in forward, per reference)
    const float* bias    = (const float*)d_in[4];
    float*       out     = (float*)d_out;

    dim3 grid(GN / BN, GM / BM);               // 32 x 64 blocks
    dim3 block(256);                           // 8 wave32
    const size_t shmem = 2 * (size_t)(BM + BN) * PAIRS * sizeof(unsigned); // 64 KB
    qlinear_wmma_kernel<<<grid, block, shmem, stream>>>(x, wpacked, scales, bias, out);
}
